// Pew_LSTM_layer_longmem_batch_8950711845911
// MI455X (gfx1250) — matrix operand order repaired and checked by reading, not compiled, not run on hardware
//
#include <hip/hip_runtime.h>

typedef __attribute__((ext_vector_type(16))) __bf16 v16bf;
typedef __attribute__((ext_vector_type(8)))  __bf16 v8bf;
typedef __attribute__((ext_vector_type(8)))  float  v8f;

namespace {
constexpr int kB = 64, kD = 90, kS = 24, kI = 256, kH = 512, kW = 32;
constexpr int kT = kD * kS;                 // 2160 timesteps
constexpr int kM = kT * kB;                 // 138240 flattened rows
constexpr size_t kOutHalf = (size_t)kB * kT * kH;  // 70,778,880 floats per output
constexpr int kScanBlocks = 32;             // one block per 16-wide N tile of H
}

__device__ __forceinline__ float fast_rcp(float x) { return __builtin_amdgcn_rcpf(x); }
__device__ __forceinline__ float sigm(float x) {
  return fast_rcp(1.0f + __expf(-x));       // v_exp_f32 + v_rcp_f32, no IEEE div chain
}
__device__ __forceinline__ float tanh_fast(float x) {
  float e = __expf(-2.0f * x);
  return (1.0f - e) * fast_rcp(1.0f + e);
}

__device__ __forceinline__ void wmma_bf16(v8f& acc, v16bf a, v16bf b) {
  // (neg_a, A, neg_b, B, c_mod, C, reuse_a, reuse_b)
  acc = __builtin_amdgcn_wmma_f32_16x16x32_bf16(false, a, false, b, (short)0, acc,
                                                false, false);
}

// ---- A fragment: 16x32 bf16 tile from row-major A (leading dim ldk elements) ----
// ISA 16-bit A layout: lanes 0-15 row M=lane, halves 0-7 = K+0..7, 8-15 = K+16..23;
// lanes 16-31 row M=lane-16, halves 0-7 = K+8..15, 8-15 = K+24..31.
__device__ __forceinline__ v16bf load_a_frag(const __bf16* __restrict__ A, int row,
                                             int ldk, int kb, int lane) {
  const __bf16* p = A + (size_t)row * ldk + kb + ((lane & 16) ? 8 : 0);
  v8bf lo = *(const v8bf*)p;
  v8bf hi = *(const v8bf*)(p + 16);
  v16bf a;
#pragma unroll
  for (int e = 0; e < 8; ++e) { a[e] = lo[e]; a[8 + e] = hi[e]; }
  return a;
}

// ---- B fragment from packed weights: lane holds N%16 (lanes 16-31 K+8..15/K+24..31), halves = K in A-fragment order ----
__device__ __forceinline__ v16bf load_b_frag(const __bf16* __restrict__ Wp, int ntile,
                                             int KT, int kt, int lane) {
  return *(const v16bf*)(Wp + ((size_t)(ntile * KT + kt) * 32 + lane) * 16);
}

// B fragment from an LDS-staged weight slab (this block's ntile only, KT=16)
__device__ __forceinline__ v16bf lds_b_frag(const __bf16* Ws, int kt, int lane) {
  return *(const v16bf*)(Ws + ((kt * 32 + lane) << 4));
}

// =====================  packing kernels  =====================

// w is (kH x K) row-major f32; dst is B-fragment-packed bf16
__global__ void pack_weight(const float* __restrict__ w, __bf16* __restrict__ dst, int K) {
  int idx = blockIdx.x * blockDim.x + threadIdx.x;
  if (idx >= kH * K) return;
  int n = idx / K, k = idx - n * K;
  int KT = K >> 5;
  int off = (((n >> 4) * KT + (k >> 5)) * 32 + (n & 15) + (((k >> 3) & 1) << 4)) * 16 + (k & 7) + (((k >> 4) & 1) << 3);
  dst[off] = (__bf16)w[idx];
}

// x_input (B,D,S,I) -> xs/hd/hw/hm as (T*B, I) bf16 with lookback shifts + masks
__global__ void pack_x_kernel(const float* __restrict__ x, __bf16* __restrict__ xs,
                              __bf16* __restrict__ hd, __bf16* __restrict__ hw,
                              __bf16* __restrict__ hm) {
  int idx = blockIdx.x * blockDim.x + threadIdx.x;
  if (idx >= kM * kI) return;
  int row = idx / kI, k = idx - row * kI;
  int t = row / kB, b = row - t * kB;
  int d = t / kS, s = t - d * kS;
  const size_t xb = (size_t)b * kT * kI;
  const __bf16 z = (__bf16)0.0f;
  xs[idx] = (__bf16)x[xb + (size_t)t * kI + k];
  hd[idx] = (d >= 1) ? (__bf16)x[xb + (size_t)(t - kS) * kI + k] : z;
  hw[idx] = (d >= 7) ? (__bf16)x[xb + (size_t)(t - 6 * kS) * kI + k] : z;
  int dm = (d + kD - 29) % kD;  // reference quirk: (d-29) mod D, masked by d>=28
  hm[idx] = (d >= 28) ? (__bf16)x[xb + ((size_t)dm * kS + s) * kI + k] : z;
}

__global__ void pack_wx_kernel(const float* __restrict__ xw, __bf16* __restrict__ wsb) {
  int idx = blockIdx.x * blockDim.x + threadIdx.x;
  if (idx >= kM * kW) return;
  int row = idx / kW, k = idx - row * kW;
  int t = row / kB, b = row - t * kB;
  wsb[idx] = (__bf16)xw[((size_t)b * kT + t) * kW + k];
}

__global__ void init_state(__bf16* __restrict__ h_bf, unsigned* __restrict__ bar) {
  int idx = blockIdx.x * blockDim.x + threadIdx.x;
  if (idx >= kB * kH) return;
  h_bf[idx] = (__bf16)0.0f;
  if (idx == 0) *bar = 0u;
}

// =====================  precompute GEMMs (parallel phase)  =====================

// e = sigmoid(ws @ w_e^T + b_e), K=32 (one WMMA per tile), stored bf16 row-major
__global__ void e_kernel(const __bf16* __restrict__ wsb, const __bf16* __restrict__ We,
                         const float* __restrict__ be, __bf16* __restrict__ e_out) {
  int wid = blockIdx.x * 8 + (threadIdx.x >> 5);
  int lane = threadIdx.x & 31;
  int nt = wid & 31, mt = wid >> 5;
  int arow = (mt << 4) + (lane & 15);
  v8f acc = {};
  v16bf a = load_a_frag(wsb, arow, kW, 0, lane);
  wmma_bf16(acc, a, load_b_frag(We, nt, 1, 0, lane));
  int n = (nt << 4) + (lane & 15);
  int mbase = (mt << 4) + ((lane & 16) ? 8 : 0);
  float bias = be[n];
#pragma unroll
  for (int r = 0; r < 8; ++r)
    e_out[(size_t)(mbase + r) * kH + n] = (__bf16)sigm(acc[r] + bias);
}

// pre_ho = hd@Wd^T + hw@Ww^T + hm@Wm^T   (raw, f32)
__global__ void pre_ho_kernel(const __bf16* __restrict__ hd, const __bf16* __restrict__ hw,
                              const __bf16* __restrict__ hm, const __bf16* __restrict__ Wd,
                              const __bf16* __restrict__ Ww, const __bf16* __restrict__ Wm,
                              float* __restrict__ pre_ho) {
  int wid = blockIdx.x * 8 + (threadIdx.x >> 5);
  int lane = threadIdx.x & 31;
  int nt = wid & 31, mt = wid >> 5;
  int arow = (mt << 4) + (lane & 15);
  v8f acc = {};
#pragma unroll 2
  for (int kt = 0; kt < 8; ++kt)
    wmma_bf16(acc, load_a_frag(hd, arow, kI, kt * 32, lane), load_b_frag(Wd, nt, 8, kt, lane));
#pragma unroll 2
  for (int kt = 0; kt < 8; ++kt)
    wmma_bf16(acc, load_a_frag(hw, arow, kI, kt * 32, lane), load_b_frag(Ww, nt, 8, kt, lane));
#pragma unroll 2
  for (int kt = 0; kt < 8; ++kt)
    wmma_bf16(acc, load_a_frag(hm, arow, kI, kt * 32, lane), load_b_frag(Wm, nt, 8, kt, lane));
  int n = (nt << 4) + (lane & 15);
  int mbase = (mt << 4) + ((lane & 16) ? 8 : 0);
#pragma unroll
  for (int r = 0; r < 8; ++r)
    pre_ho[(size_t)(mbase + r) * kH + n] = acc[r];
}

// pre_i = x@Wix + e@Wie + bi ; pre_g = x@Wgx + bg ; pre_f = x@Wfx + e@Wfe + bf ;
// pre_o = x@Wox + e@Woe + bo     (one pass; A fragments shared by 3-4 WMMAs)
__global__ void pre_gates_kernel(const __bf16* __restrict__ xs, const __bf16* __restrict__ e,
                                 const __bf16* __restrict__ Wix, const __bf16* __restrict__ Wgx,
                                 const __bf16* __restrict__ Wfx, const __bf16* __restrict__ Wox,
                                 const __bf16* __restrict__ Wie, const __bf16* __restrict__ Wfe,
                                 const __bf16* __restrict__ Woe,
                                 const float* __restrict__ bi, const float* __restrict__ bg,
                                 const float* __restrict__ bff, const float* __restrict__ bo,
                                 float* __restrict__ pre_i, float* __restrict__ pre_g,
                                 float* __restrict__ pre_f, float* __restrict__ pre_o) {
  int wid = blockIdx.x * 8 + (threadIdx.x >> 5);
  int lane = threadIdx.x & 31;
  int nt = wid & 31, mt = wid >> 5;
  int arow = (mt << 4) + (lane & 15);
  v8f ai = {}, ag = {}, af = {}, ao = {};
#pragma unroll 2
  for (int kt = 0; kt < 8; ++kt) {            // x part, K=256
    v16bf a = load_a_frag(xs, arow, kI, kt * 32, lane);
    wmma_bf16(ai, a, load_b_frag(Wix, nt, 8, kt, lane));
    wmma_bf16(ag, a, load_b_frag(Wgx, nt, 8, kt, lane));
    wmma_bf16(af, a, load_b_frag(Wfx, nt, 8, kt, lane));
    wmma_bf16(ao, a, load_b_frag(Wox, nt, 8, kt, lane));
  }
#pragma unroll 2
  for (int kt = 0; kt < 16; ++kt) {           // e part, K=512
    v16bf a = load_a_frag(e, arow, kH, kt * 32, lane);
    wmma_bf16(ai, a, load_b_frag(Wie, nt, 16, kt, lane));
    wmma_bf16(af, a, load_b_frag(Wfe, nt, 16, kt, lane));
    wmma_bf16(ao, a, load_b_frag(Woe, nt, 16, kt, lane));
  }
  int n = (nt << 4) + (lane & 15);
  int mbase = (mt << 4) + ((lane & 16) ? 8 : 0);
  float vi = bi[n], vg = bg[n], vf = bff[n], vo = bo[n];
#pragma unroll
  for (int r = 0; r < 8; ++r) {
    size_t idx = (size_t)(mbase + r) * kH + n;
    pre_i[idx] = ai[r] + vi;
    pre_g[idx] = ag[r] + vg;
    pre_f[idx] = af[r] + vf;
    pre_o[idx] = ao[r] + vo;
  }
}

// =====================  persistent sequential scan  =====================

// Device-wide split barrier over kScanBlocks persistent blocks.
__device__ __forceinline__ void grid_barrier(unsigned* bar, unsigned target) {
  __threadfence();   // agent-scope release of this thread's global writes
  __syncthreads();
  if (threadIdx.x == 0) {
    __hip_atomic_fetch_add(bar, 1u, __ATOMIC_RELAXED, __HIP_MEMORY_SCOPE_AGENT);
    while (__hip_atomic_load(bar, __ATOMIC_RELAXED, __HIP_MEMORY_SCOPE_AGENT) < target)
      __builtin_amdgcn_s_sleep(1);
  }
  __syncthreads();
  __threadfence();   // agent-scope acquire before reading peers' writes
}

// One persistent kernel runs all 2160 steps. Block = one N tile (nt = blockIdx.x),
// 4 waves = 4 M tiles of the batch. c and o gates live in registers for the whole
// scan; the 5 recurrent weight slabs for this nt are staged once into LDS (80 KB).
__global__ void scan_kernel(const __bf16* __restrict__ Wt, const __bf16* __restrict__ Woh,
                            const __bf16* __restrict__ Wih, const __bf16* __restrict__ Wgh,
                            const __bf16* __restrict__ Wfo,
                            const float* __restrict__ pre_ho, const float* __restrict__ pre_i,
                            const float* __restrict__ pre_g, const float* __restrict__ pre_f,
                            const float* __restrict__ pre_o,
                            __bf16* __restrict__ h_bf, __bf16* __restrict__ ho_bf,
                            unsigned* __restrict__ bar, float* __restrict__ out) {
  extern __shared__ __bf16 smem[];  // 5 slabs x 8192 bf16 (16 KB) = 80 KB
  const int nt = blockIdx.x;                 // 0..31
  const int mt = threadIdx.x >> 5;           // 0..3
  const int lane = threadIdx.x & 31;
  __bf16* sWt = smem;
  __bf16* sWoh = smem + 8192;
  __bf16* sWih = smem + 2 * 8192;
  __bf16* sWgh = smem + 3 * 8192;
  __bf16* sWfo = smem + 4 * 8192;

  {  // stage this block's weight slabs (contiguous 16 KB each in packed layout)
    const size_t wb = (size_t)nt * 8192;     // elements; KT=16 -> nt*16*512
    const uint4* s0 = (const uint4*)(Wt + wb);
    const uint4* s1 = (const uint4*)(Woh + wb);
    const uint4* s2 = (const uint4*)(Wih + wb);
    const uint4* s3 = (const uint4*)(Wgh + wb);
    const uint4* s4 = (const uint4*)(Wfo + wb);
    uint4* d0 = (uint4*)sWt;  uint4* d1 = (uint4*)sWoh; uint4* d2 = (uint4*)sWih;
    uint4* d3 = (uint4*)sWgh; uint4* d4 = (uint4*)sWfo;
    for (int i = threadIdx.x; i < 1024; i += 128) {
      d0[i] = s0[i]; d1[i] = s1[i]; d2[i] = s2[i]; d3[i] = s3[i]; d4[i] = s4[i];
    }
  }
  __syncthreads();

  const int arow = (mt << 4) + (lane & 15);
  const int n = (nt << 4) + (lane & 15);
  const int mbase = (mt << 4) + ((lane & 16) ? 8 : 0);
  v8f c_reg = {};
  v8f o_reg = {};
  unsigned target = 0;

  for (int t = 0; t < kT; ++t) {
    // ---- phase A: h_o = sigm(pre_ho + 2*h@Wt^T), o = sigm(pre_o + h@Woh^T) ----
    v8f at = {}, ao = {};
#pragma unroll 4
    for (int kt = 0; kt < 16; ++kt) {
      v16bf a = load_a_frag(h_bf, arow, kH, kt * 32, lane);
      wmma_bf16(at, a, lds_b_frag(sWt, kt, lane));
      wmma_bf16(ao, a, lds_b_frag(sWoh, kt, lane));
    }
    {
      size_t idx0 = ((size_t)t * kB + mbase) * kH + n;
      if (t + 1 < kT) {  // warm L2 for next step's activation rows
        size_t nx = ((size_t)(t + 1) * kB + mbase) * kH + n;
        __builtin_prefetch(pre_ho + nx, 0, 1);
        __builtin_prefetch(pre_i + nx, 0, 1);
        __builtin_prefetch(pre_f + nx, 0, 1);
      }
#pragma unroll
      for (int r = 0; r < 8; ++r) {
        size_t idx = idx0 + (size_t)r * kH;
        ho_bf[(mbase + r) * kH + n] = (__bf16)sigm(pre_ho[idx] + 2.0f * at[r]);
        o_reg[r] = sigm(pre_o[idx] + ao[r]);
      }
    }
    target += kScanBlocks;
    grid_barrier(bar, target);

    // ---- phase B: i/g/f from h_o, state update, outputs ----
    v8f ai = {}, ag = {}, af = {};
#pragma unroll 4
    for (int kt = 0; kt < 16; ++kt) {
      v16bf a = load_a_frag(ho_bf, arow, kH, kt * 32, lane);
      wmma_bf16(ai, a, lds_b_frag(sWih, kt, lane));
      wmma_bf16(ag, a, lds_b_frag(sWgh, kt, lane));
      wmma_bf16(af, a, lds_b_frag(sWfo, kt, lane));
    }
    {
      size_t idx0 = ((size_t)t * kB + mbase) * kH + n;
#pragma unroll
      for (int r = 0; r < 8; ++r) {
        int m = mbase + r;
        size_t idx = idx0 + (size_t)r * kH;
        float ig = sigm(pre_i[idx] + ai[r]);
        float g = tanh_fast(pre_g[idx] + ag[r]);
        float f = sigm(pre_f[idx] + af[r]);
        c_reg[r] = f * c_reg[r] + ig * g;
        float h = o_reg[r] * tanh_fast(c_reg[r]);
        h_bf[m * kH + n] = (__bf16)h;
        size_t obase = ((size_t)m * kT + t) * kH + n;
        out[obase] = h;
        out[kOutHalf + obase] = c_reg[r];
      }
    }
    target += kScanBlocks;
    grid_barrier(bar, target);
  }
}

// =====================  host launcher  =====================

extern "C" void kernel_launch(void* const* d_in, const int* in_sizes, int n_in,
                              void* d_out, int out_size, void* d_ws, size_t ws_size,
                              hipStream_t stream) {
  (void)in_sizes; (void)n_in; (void)out_size; (void)ws_size;
  const float* x_input = (const float*)d_in[0];
  const float* x_weather = (const float*)d_in[1];
  const float* w_ix = (const float*)d_in[2];
  const float* w_ih = (const float*)d_in[3];
  const float* w_ie = (const float*)d_in[4];
  const float* b_i = (const float*)d_in[5];
  const float* w_fx = (const float*)d_in[6];
  const float* w_fo = (const float*)d_in[7];
  const float* w_fe = (const float*)d_in[8];
  const float* b_f = (const float*)d_in[9];
  const float* w_ox = (const float*)d_in[10];
  const float* w_oh = (const float*)d_in[11];
  const float* w_oe = (const float*)d_in[12];
  const float* b_o = (const float*)d_in[13];
  const float* w_gx = (const float*)d_in[14];
  const float* w_gh = (const float*)d_in[15];
  const float* b_g = (const float*)d_in[16];
  const float* w_d = (const float*)d_in[17];
  const float* w_w = (const float*)d_in[18];
  const float* w_m = (const float*)d_in[19];
  const float* w_t = (const float*)d_in[20];
  const float* w_e = (const float*)d_in[21];
  const float* b_e = (const float*)d_in[22];
  float* out = (float*)d_out;

  char* p = (char*)d_ws;
  auto alloc = [&](size_t bytes) -> void* {
    void* r = (void*)p;
    p += (bytes + 255) & ~(size_t)255;
    return r;
  };
  const size_t wHH = (size_t)kH * kH * sizeof(__bf16);
  const size_t wHI = (size_t)kH * kI * sizeof(__bf16);
  __bf16* Wt_p  = (__bf16*)alloc(wHH);
  __bf16* Woh_p = (__bf16*)alloc(wHH);
  __bf16* Wih_p = (__bf16*)alloc(wHH);
  __bf16* Wgh_p = (__bf16*)alloc(wHH);
  __bf16* Wfo_p = (__bf16*)alloc(wHH);
  __bf16* Wie_p = (__bf16*)alloc(wHH);
  __bf16* Wfe_p = (__bf16*)alloc(wHH);
  __bf16* Woe_p = (__bf16*)alloc(wHH);
  __bf16* Wix_p = (__bf16*)alloc(wHI);
  __bf16* Wgx_p = (__bf16*)alloc(wHI);
  __bf16* Wfx_p = (__bf16*)alloc(wHI);
  __bf16* Wox_p = (__bf16*)alloc(wHI);
  __bf16* Wd_p  = (__bf16*)alloc(wHI);
  __bf16* Ww_p  = (__bf16*)alloc(wHI);
  __bf16* Wm_p  = (__bf16*)alloc(wHI);
  __bf16* We_p  = (__bf16*)alloc((size_t)kH * kW * sizeof(__bf16));
  __bf16* xs  = (__bf16*)alloc((size_t)kM * kI * sizeof(__bf16));
  __bf16* hd  = (__bf16*)alloc((size_t)kM * kI * sizeof(__bf16));
  __bf16* hw  = (__bf16*)alloc((size_t)kM * kI * sizeof(__bf16));
  __bf16* hm  = (__bf16*)alloc((size_t)kM * kI * sizeof(__bf16));
  __bf16* wsb = (__bf16*)alloc((size_t)kM * kW * sizeof(__bf16));
  __bf16* e_b = (__bf16*)alloc((size_t)kM * kH * sizeof(__bf16));
  float* pre_ho = (float*)alloc((size_t)kM * kH * sizeof(float));
  float* pre_i  = (float*)alloc((size_t)kM * kH * sizeof(float));
  float* pre_g  = (float*)alloc((size_t)kM * kH * sizeof(float));
  float* pre_f  = (float*)alloc((size_t)kM * kH * sizeof(float));
  float* pre_o  = (float*)alloc((size_t)kM * kH * sizeof(float));
  __bf16* h_bf  = (__bf16*)alloc((size_t)kB * kH * sizeof(__bf16));
  __bf16* ho_bf = (__bf16*)alloc((size_t)kB * kH * sizeof(__bf16));
  unsigned* bar = (unsigned*)alloc(256);

  // ---- phase 0: pack weights + inputs to bf16 WMMA layouts ----
  const int gHH = (kH * kH + 255) / 256, gHI = (kH * kI + 255) / 256;
  pack_weight<<<gHH, 256, 0, stream>>>(w_t, Wt_p, kH);
  pack_weight<<<gHH, 256, 0, stream>>>(w_oh, Woh_p, kH);
  pack_weight<<<gHH, 256, 0, stream>>>(w_ih, Wih_p, kH);
  pack_weight<<<gHH, 256, 0, stream>>>(w_gh, Wgh_p, kH);
  pack_weight<<<gHH, 256, 0, stream>>>(w_fo, Wfo_p, kH);
  pack_weight<<<gHH, 256, 0, stream>>>(w_ie, Wie_p, kH);
  pack_weight<<<gHH, 256, 0, stream>>>(w_fe, Wfe_p, kH);
  pack_weight<<<gHH, 256, 0, stream>>>(w_oe, Woe_p, kH);
  pack_weight<<<gHI, 256, 0, stream>>>(w_ix, Wix_p, kI);
  pack_weight<<<gHI, 256, 0, stream>>>(w_gx, Wgx_p, kI);
  pack_weight<<<gHI, 256, 0, stream>>>(w_fx, Wfx_p, kI);
  pack_weight<<<gHI, 256, 0, stream>>>(w_ox, Wox_p, kI);
  pack_weight<<<gHI, 256, 0, stream>>>(w_d, Wd_p, kI);
  pack_weight<<<gHI, 256, 0, stream>>>(w_w, Ww_p, kI);
  pack_weight<<<gHI, 256, 0, stream>>>(w_m, Wm_p, kI);
  pack_weight<<<(kH * kW + 255) / 256, 256, 0, stream>>>(w_e, We_p, kW);

  pack_x_kernel<<<(kM * kI + 255) / 256, 256, 0, stream>>>(x_input, xs, hd, hw, hm);
  pack_wx_kernel<<<(kM * kW + 255) / 256, 256, 0, stream>>>(x_weather, wsb);
  init_state<<<(kB * kH + 255) / 256, 256, 0, stream>>>(h_bf, bar);

  // ---- phase 1: parallel precompute GEMMs (one wave per 16x16 tile) ----
  const int bigGrid = (kM / 16) * (kH / 16) / 8;  // 34560 blocks x 8 waves
  e_kernel<<<bigGrid, 256, 0, stream>>>(wsb, We_p, b_e, e_b);
  pre_ho_kernel<<<bigGrid, 256, 0, stream>>>(hd, hw, hm, Wd_p, Ww_p, Wm_p, pre_ho);
  pre_gates_kernel<<<bigGrid, 256, 0, stream>>>(xs, e_b, Wix_p, Wgx_p, Wfx_p, Wox_p,
                                                Wie_p, Wfe_p, Woe_p, b_i, b_g, b_f, b_o,
                                                pre_i, pre_g, pre_f, pre_o);

  // ---- phase 2: one persistent kernel runs the whole 2160-step scan ----
  scan_kernel<<<kScanBlocks, 128, 5 * 8192 * sizeof(__bf16), stream>>>(
      Wt_p, Woh_p, Wih_p, Wgh_p, Wfo_p, pre_ho, pre_i, pre_g, pre_f, pre_o,
      h_bf, ho_bf, bar, out);
}
